// Net_70798240907740
// MI455X (gfx1250) — matrix operand order repaired and checked by reading, not compiled, not run on hardware
//
#include <hip/hip_runtime.h>
#include <cstdint>
#include <cstddef>

// ---------------------------------------------------------------------------
// Darknet-19 forward (batch 2, 416x416) for MI455X (gfx1250, wave32, WMMA).
// - NHWC bf16 activations: B fragments and C stores are contiguous vector ops
// - Weights pre-packed per launch into exact WMMA A-fragment layout (bf16)
// - Conv = implicit GEMM on v_wmma_f32_16x16x32_bf16
//   one wave32 = 64 oc x 32 px: per K-block 2 B loads + 4 A loads + 8 WMMAs
// - Zero guard page handles padding: no per-element selects in the hot loop
// ---------------------------------------------------------------------------

typedef __attribute__((ext_vector_type(16))) __bf16          v16bf;
typedef __attribute__((ext_vector_type(8)))  float           v8f;
typedef __attribute__((ext_vector_type(16))) unsigned short  v16u;
typedef __attribute__((ext_vector_type(8)))  unsigned short  v8us;
typedef __attribute__((ext_vector_type(8)))  unsigned int    v8u;

static __device__ __forceinline__ unsigned short f32_to_bf16_rne(float f) {
    union { float f; unsigned int u; } v; v.f = f;
    unsigned int u = v.u;
    u += 0x7FFFu + ((u >> 16) & 1u);
    return (unsigned short)(u >> 16);
}
static __device__ __forceinline__ float bf16_to_f32(unsigned short u) {
    union { unsigned int i; float f; } v; v.i = ((unsigned int)u) << 16;
    return v.f;
}

static __device__ __forceinline__ v8f wmma_bf16(v16u au, v16u bu, v8f acc) {
    v16bf a = __builtin_bit_cast(v16bf, au);
    v16bf b = __builtin_bit_cast(v16bf, bu);
    return __builtin_amdgcn_wmma_f32_16x16x32_bf16(
        false, a, false, b, (short)0, acc, false, false);
}

// A-fragment (16x32 bf16): element 2i(+1) of lane (half,m) holds
//   K = (i<4?0:16) + half*8 + (i&3)*2 (+1)
static __device__ __forceinline__ int a_kidx(int i, int half) {
    return ((i < 4) ? 0 : 16) + half * 8 + ((i & 3) << 1);
}
// B-fragment order chosen here (32x16 bf16): element e of lane holds K = half*16 + e, N = lane&15
// => one lane's 16 elements are 16 CONSECUTIVE K values (contiguous in NHWC).

#define MT 4   // out-channel subtiles per wave (64 oc)
#define NT 2   // pixel subtiles per wave (32 px)

// ---------------------------------------------------------------------------
// Weight pack: fp32 OIHW -> bf16 WMMA A-fragments, tap-major K order
// (k enumerated as (tap, c) so it matches NHWC B's contiguous channels).
// ---------------------------------------------------------------------------
__global__ __launch_bounds__(256) void pack_weights_kernel(
    const float* __restrict__ Wt, unsigned short* __restrict__ out,
    int K, int C, int R, int CK, int RS, int nKB, int nCB, long total)
{
    long idx = (long)blockIdx.x * blockDim.x + threadIdx.x;
    if (idx >= total) return;
    const int elt  = (int)(idx & 15);
    const int lane = (int)((idx >> 4) & 31);
    const long t   = idx >> 9;
    const int kb    = (int)(t % nKB);
    const int mtile = (int)(t / nKB);

    const int half = lane >> 4, mr = lane & 15;
    const int i = elt >> 1, d = elt & 1;
    const int kl = half * 16 + elt;
    const int row = mtile * 16 + mr;

    float v = 0.0f;
    if (row < K) {
        if (CK <= 32) {                          // flat (layer 1)
            if (kl < CK) v = Wt[(size_t)row * CK + kl];
        } else {                                 // tap-major, C % 32 == 0
            const int tap  = kb / nCB;
            const int cblk = kb - tap * nCB;
            const int c    = cblk * 32 + kl;
            v = Wt[(size_t)row * CK + (size_t)c * RS + tap];
        }
    }
    out[idx] = f32_to_bf16_rne(v);
}

// Network input: NCHW fp32 -> NHWC bf16
__global__ __launch_bounds__(256) void input_to_nhwc_bf16_kernel(
    const float* __restrict__ in, unsigned short* __restrict__ out,
    int C, int HW, long n)
{
    long i = (long)blockIdx.x * blockDim.x + threadIdx.x;
    if (i >= n) return;
    const int  c  = (int)(i % C);
    const long p  = i / C;                 // (nb, y, x) flat
    const long nb = p / HW;
    const long rm = p - nb * HW;
    out[i] = f32_to_bf16_rne(in[(nb * C + c) * HW + rm]);
}

// ---------------------------------------------------------------------------
// Conv (NHWC bf16): one wave32 computes 64(oc) x 32(pixel).
// ---------------------------------------------------------------------------
__global__ __launch_bounds__(256) void conv_wmma_kernel(
    const unsigned short* __restrict__ X,   // bf16 [N, H, W, C]
    const unsigned short* __restrict__ Ap,  // packed bf16 A fragments
    const unsigned short* __restrict__ ZP,  // zero guard page (>= 2*C bytes)
    unsigned short* __restrict__ Y,         // bf16 [N, H, W, K]
    int N, int C, int H, int W, int K, int R, int pad, int nKB)
{
    const int HW     = H * W;
    const int NPIX   = N * HW;
    const int RS     = R * R;
    const int CK     = C * RS;
    const int tilesN = (NPIX + 16 * NT - 1) / (16 * NT);
    const int tilesM = (K + 16 * MT - 1) / (16 * MT);

    const int wave = blockIdx.x * (blockDim.x >> 5) + (threadIdx.x >> 5);
    if (wave >= tilesM * tilesN) return;       // wave-uniform: EXEC stays full

    const int tm = wave / tilesN;
    const int tn = wave - tm * tilesN;

    const int lane = threadIdx.x & 31;
    const int half = lane >> 4;
    const int mr   = lane & 15;

    const int ocb = tm * (16 * MT);
    const int pnb = tn * (16 * NT);

    // Pixel decode per pixel-subtile (N index of subtile nt is mr)
    int  pixv[NT], nbv[NT], oyv[NT], oxv[NT];
    bool pval[NT];
    #pragma unroll
    for (int nt = 0; nt < NT; ++nt) {
        const int pix = pnb + nt * 16 + mr;
        pval[nt] = pix < NPIX;
        const int pc  = pval[nt] ? pix : (NPIX - 1);
        pixv[nt] = pix;
        nbv[nt]  = pc / HW;
        const int rem = pc - nbv[nt] * HW;
        oyv[nt] = rem / W;
        oxv[nt] = rem - oyv[nt] * W;
    }

    // Packed-A base per oc-subtile; step per kb is 512 elements (1 KB).
    const unsigned short* apB[MT];
    #pragma unroll
    for (int mt = 0; mt < MT; ++mt)
        apB[mt] = Ap + ((size_t)(tm * MT + mt) * nKB * 512) + (size_t)lane * 16;

    v8f acc[MT][NT];
    #pragma unroll
    for (int mt = 0; mt < MT; ++mt)
        #pragma unroll
        for (int nt = 0; nt < NT; ++nt) acc[mt][nt] = (v8f){};

    if (CK <= 32) {
        // ---- Flat path (layer 1, C=3): one K block, scalar im2col gather.
        v16u bu[NT];
        #pragma unroll
        for (int nt = 0; nt < NT; ++nt) {
            #pragma unroll
            for (int e = 0; e < 16; ++e) {
                const int ck = half * 16 + e;   // B element e holds K = half*16+e
                unsigned short v = 0;
                if (pval[nt] && ck < CK) {
                    int c  = ck / RS;
                    int tt = ck - c * RS;
                    int ky = tt / R;
                    int kx = tt - ky * R;
                    int iy = oyv[nt] + ky - pad;
                    int ix = oxv[nt] + kx - pad;
                    if ((unsigned)iy < (unsigned)H && (unsigned)ix < (unsigned)W)
                        v = X[(((size_t)nbv[nt] * H + iy) * W + ix) * C + c];
                }
                bu[nt][e] = v;
            }
        }
        #pragma unroll
        for (int mt = 0; mt < MT; ++mt) {
            v16u au = __builtin_bit_cast(v16u, *(const v8u*)apB[mt]);
            #pragma unroll
            for (int nt = 0; nt < NT; ++nt)
                acc[mt][nt] = wmma_bf16(au, bu[nt], acc[mt][nt]);
        }
    } else {
        // ---- Hot path: everything is contiguous 32B vector loads.
        const int nCB = C >> 5;                // C % 32 == 0 here
        for (int ky = 0; ky < R; ++ky) {
            for (int kx = 0; kx < R; ++kx) {
                const int tap = ky * R + kx;

                const unsigned short* Bp[NT];
                #pragma unroll
                for (int nt = 0; nt < NT; ++nt) {
                    const int iy = oyv[nt] + ky - pad;
                    const int ix = oxv[nt] + kx - pad;
                    const bool vB = pval[nt] &&
                        (unsigned)iy < (unsigned)H && (unsigned)ix < (unsigned)W;
                    // Invalid lanes read the zero page: no selects in the loop.
                    Bp[nt] = vB
                        ? X + (((size_t)nbv[nt] * H + iy) * W + ix) * C + half * 16
                        : ZP + half * 16;
                }

                const unsigned short* ap0 = apB[0] + (size_t)(tap * nCB) * 512;
                const unsigned short* ap1 = apB[1] + (size_t)(tap * nCB) * 512;
                const unsigned short* ap2 = apB[2] + (size_t)(tap * nCB) * 512;
                const unsigned short* ap3 = apB[3] + (size_t)(tap * nCB) * 512;

                for (int cb = 0; cb < nCB; ++cb) {
                    const v16u b0 = __builtin_bit_cast(v16u, *(const v8u*)(Bp[0] + (cb << 5)));
                    const v16u b1 = __builtin_bit_cast(v16u, *(const v8u*)(Bp[1] + (cb << 5)));
                    const size_t ko = (size_t)cb * 512;

                    v16u a;
                    a = __builtin_bit_cast(v16u, *(const v8u*)(ap0 + ko));
                    acc[0][0] = wmma_bf16(a, b0, acc[0][0]);
                    acc[0][1] = wmma_bf16(a, b1, acc[0][1]);
                    a = __builtin_bit_cast(v16u, *(const v8u*)(ap1 + ko));
                    acc[1][0] = wmma_bf16(a, b0, acc[1][0]);
                    acc[1][1] = wmma_bf16(a, b1, acc[1][1]);
                    a = __builtin_bit_cast(v16u, *(const v8u*)(ap2 + ko));
                    acc[2][0] = wmma_bf16(a, b0, acc[2][0]);
                    acc[2][1] = wmma_bf16(a, b1, acc[2][1]);
                    a = __builtin_bit_cast(v16u, *(const v8u*)(ap3 + ko));
                    acc[3][0] = wmma_bf16(a, b0, acc[3][0]);
                    acc[3][1] = wmma_bf16(a, b1, acc[3][1]);
                }
            }
        }
    }

    // Store NHWC bf16. Lane holds (M = i + 8*half, N = mr) in acc VGPR i:
    // 8 consecutive output channels at one pixel -> one 16B store per subtile.
    #pragma unroll
    for (int nt = 0; nt < NT; ++nt) {
        if (pval[nt]) {
            unsigned short* yp = Y + (size_t)pixv[nt] * K + ocb + half * 8;
            #pragma unroll
            for (int mt = 0; mt < MT; ++mt) {
                if (ocb + (mt + 1) * 16 <= K) {          // wave-uniform
                    v8us o;
                    #pragma unroll
                    for (int i = 0; i < 8; ++i) o[i] = f32_to_bf16_rne(acc[mt][nt][i]);
                    *(v8us*)(yp + mt * 16) = o;
                } else {                                  // ragged tail (layer 19)
                    #pragma unroll
                    for (int i = 0; i < 8; ++i) {
                        const int oc = ocb + mt * 16 + half * 8 + i;
                        if (oc < K)
                            Y[(size_t)pixv[nt] * K + oc] = f32_to_bf16_rne(acc[mt][nt][i]);
                    }
                }
            }
        }
    }
}

// 2x2 stride-2 maxpool over NHWC bf16
__global__ __launch_bounds__(256) void maxpool2_kernel(
    const unsigned short* __restrict__ X, unsigned short* __restrict__ Y,
    int C, int H, int W, long total)   // total = N*OH*OW*C
{
    const int OH = H >> 1, OW = W >> 1;
    for (long i = (long)blockIdx.x * blockDim.x + threadIdx.x; i < total;
         i += (long)gridDim.x * blockDim.x) {
        const int  c  = (int)(i % C);
        const long q  = i / C;
        const int  ox = (int)(q % OW);
        const long r  = q / OW;
        const int  oy = (int)(r % OH);
        const long nb = r / OH;
        const unsigned short* p =
            X + (((nb * H + (size_t)oy * 2) * W + (size_t)ox * 2) * C) + c;
        const size_t sx = C, sy = (size_t)W * C;
        float m = fmaxf(fmaxf(bf16_to_f32(p[0]),  bf16_to_f32(p[sx])),
                        fmaxf(bf16_to_f32(p[sy]), bf16_to_f32(p[sy + sx])));
        Y[i] = f32_to_bf16_rne(m);   // exact: m is one of the bf16 inputs
    }
}

// Global average pool (fp32 accumulate) + softmax. One block per batch image.
__global__ __launch_bounds__(256) void avgpool_softmax_kernel(
    const unsigned short* __restrict__ X,  // bf16 NHWC [N, HW, C]
    float* __restrict__ out,               // fp32 [N, C]
    int C, int HW)
{
    __shared__ float logits[1024];
    __shared__ float red[256];

    const int b   = blockIdx.x;
    const int tid = threadIdx.x;
    const unsigned short* xb = X + (size_t)b * HW * C;
    const float inv = 1.0f / (float)HW;

    for (int c = tid; c < C; c += blockDim.x) {
        float s = 0.0f;
        for (int i = 0; i < HW; ++i) s += bf16_to_f32(xb[(size_t)i * C + c]);
        logits[c] = s * inv;
    }
    __syncthreads();

    float m = -3.4e38f;
    for (int c = tid; c < C; c += blockDim.x) m = fmaxf(m, logits[c]);
    red[tid] = m;
    __syncthreads();
    for (int s = 128; s > 0; s >>= 1) {
        if (tid < s) red[tid] = fmaxf(red[tid], red[tid + s]);
        __syncthreads();
    }
    m = red[0];
    __syncthreads();

    float sum = 0.0f;
    for (int c = tid; c < C; c += blockDim.x) {
        float e = expf(logits[c] - m);
        logits[c] = e;
        sum += e;
    }
    red[tid] = sum;
    __syncthreads();
    for (int s = 128; s > 0; s >>= 1) {
        if (tid < s) red[tid] += red[tid + s];
        __syncthreads();
    }
    const float rs = 1.0f / red[0];

    for (int c = tid; c < C; c += blockDim.x)
        out[(size_t)b * C + c] = logits[c] * rs;
}

// ---------------------------------------------------------------------------

extern "C" void kernel_launch(void* const* d_in, const int* in_sizes, int n_in,
                              void* d_out, int out_size, void* d_ws, size_t ws_size,
                              hipStream_t stream) {
    (void)in_sizes; (void)n_in; (void)out_size; (void)ws_size;

    const float* x = (const float*)d_in[0];
    const float* wts[19];
    for (int i = 0; i < 19; ++i) wts[i] = (const float*)d_in[5 + i];

    static const int  plan_oc[19]   = {32, 64, 128, 64, 128, 256, 128, 256,
                                       512, 256, 512, 256, 512,
                                       1024, 512, 1024, 512, 1024, 1000};
    static const int  plan_k[19]    = {3, 3, 3, 1, 3, 3, 1, 3,
                                       3, 1, 3, 1, 3,
                                       3, 1, 3, 1, 3, 1};
    static const bool pool_after[19]= {true, true, false, false, true, false, false, true,
                                       false, false, false, false, true,
                                       false, false, false, false, false, false};

    // ---- workspace layout (bf16 NHWC activations) ----
    // r0 <= 22.2 MB, r1 <= 5.6 MB, x_bf16 2.1 MB, zero page 4 KB, packed A ~41 MB
    unsigned short* r0  = (unsigned short*)d_ws;
    unsigned short* r1  = (unsigned short*)((char*)d_ws + ((size_t)24 << 20));
    unsigned short* xbf = (unsigned short*)((char*)d_ws + ((size_t)36 << 20));
    unsigned short* zp  = (unsigned short*)((char*)d_ws + ((size_t)39 << 20));
    unsigned short* apk = (unsigned short*)((char*)d_ws + ((size_t)40 << 20));

    const int N = 2;

    // ---- 0) zero guard page (capturable async memset) ----
    hipMemsetAsync(zp, 0, 4096, stream);

    // ---- 1) pack weights into WMMA A-fragment layout ----
    size_t apOff[19];
    int    nKBv[19];
    {
        int Cc = 3;
        size_t off = 0;
        for (int i = 0; i < 19; ++i) {
            const int K = plan_oc[i], R = plan_k[i];
            const int RS = R * R, CK = Cc * RS;
            const int nCB = (CK <= 32) ? 1 : (Cc >> 5);
            const int nKB = (CK <= 32) ? 1 : RS * nCB;
            const int nMtile = ((K + 16 * MT - 1) / (16 * MT)) * MT;
            const long total = (long)nMtile * nKB * 512;

            apOff[i] = off;
            nKBv[i]  = nKB;
            pack_weights_kernel<<<(int)((total + 255) / 256), 256, 0, stream>>>(
                wts[i], apk + off, K, Cc, R, CK, RS, nKB, nCB, total);
            off += (size_t)total;
            Cc = K;
        }
    }

    // ---- 2) network input NCHW fp32 -> NHWC bf16 ----
    {
        const long n = (long)N * 3 * 416 * 416;
        input_to_nhwc_bf16_kernel<<<(int)((n + 255) / 256), 256, 0, stream>>>(
            x, xbf, 3, 416 * 416, n);
    }

    // ---- 3) conv / pool pipeline (NHWC bf16) ----
    int H = 416, W = 416, Cc = 3;
    const unsigned short* cur = xbf;
    int step = 0;
    auto nextbuf = [&]() -> unsigned short* {
        unsigned short* p = ((step & 1) == 0) ? r0 : r1; ++step; return p;
    };

    for (int i = 0; i < 19; ++i) {
        const int K = plan_oc[i];
        const int R = plan_k[i];
        const int pad = (R - 1) / 2;
        unsigned short* out = nextbuf();

        const int NPIX   = N * H * W;
        const long tiles = (long)((K + 16 * MT - 1) / (16 * MT))
                         * (long)((NPIX + 16 * NT - 1) / (16 * NT));
        const int grid = (int)((tiles + 7) / 8);           // 8 wave32 per block

        conv_wmma_kernel<<<grid, 256, 0, stream>>>(
            cur, apk + apOff[i], zp, out, N, Cc, H, W, K, R, pad, nKBv[i]);
        cur = out;
        Cc  = K;

        if (pool_after[i]) {
            unsigned short* pout = nextbuf();
            const long total = (long)N * (H / 2) * (W / 2) * Cc;
            int pgrid = (int)((total + 255) / 256);
            if (pgrid > 4096) pgrid = 4096;
            maxpool2_kernel<<<pgrid, 256, 0, stream>>>(cur, pout, Cc, H, W, total);
            H >>= 1; W >>= 1;
            cur = pout;
        }
    }

    // ---- 4) head: global avg pool + softmax -> d_out [2, 1000] fp32 ----
    avgpool_softmax_kernel<<<N, 256, 0, stream>>>(cur, (float*)d_out, 1000, H * W);
}
